// DecoderAblationAttModule_7636451852736
// MI455X (gfx1250) — compile-verified, matrix operand order checked by reading, not run on hardware
//
#include <hip/hip_runtime.h>
#include <hip/hip_bf16.h>

// ---------------------------------------------------------------------------
// Problem constants (match reference)
// ---------------------------------------------------------------------------
constexpr int B_  = 128;    // batch
constexpr int R_  = 36;     // regions
constexpr int F_  = 2048;   // feature dim
constexpr int D_  = 1024;   // hidden dim
constexpr int A_  = 512;    // attention dim
constexpr int E_  = 1024;   // embed dim
constexpr int V_  = 10000;  // vocab
constexpr int ML_ = 21;     // max len
constexpr int T_  = 20;     // decode steps
constexpr int KG_ = F_ + E_ + D_;   // 4096 : K of fused gates GEMM  [aw|emb|h]
constexpr int KP_ = D_ + F_;        // 3072 : K of fused pred GEMM   [h|aw]
constexpr int NG_ = 4 * D_;         // 4096 : N of gates GEMM

typedef __bf16 bf16_t;
typedef __attribute__((ext_vector_type(16))) __bf16 v16bf;
typedef __attribute__((ext_vector_type(8)))  __bf16 bf16x8;
typedef __attribute__((ext_vector_type(8)))  float  v8f;

__device__ __forceinline__ float sigmoidf_(float x) { return 1.0f / (1.0f + __expf(-x)); }

// ---------------------------------------------------------------------------
// Fragment load for 16-bit WMMA operands (ISA 7.12.2, 16-bit A 16x32):
//   lanes 0-15 : row = lane,    elems 0..7 -> K 0..7,  elems 8..15 -> K 16..23
//   lanes16-31 : row = lane-16, elems 0..7 -> K 8..15, elems 8..15 -> K 24..31
// Caller pre-offsets the pointer by (row*ld + ((lane&16)?8:0)).
// Two 16B loads -> global_load_b128 x2.
// ---------------------------------------------------------------------------
__device__ __forceinline__ v16bf load_frag(const bf16_t* p) {
  bf16x8 lo = *(const bf16x8*)(p);
  bf16x8 hi = *(const bf16x8*)(p + 16);
  v16bf r;
#pragma unroll
  for (int i = 0; i < 8; ++i) { r[i] = lo[i]; r[i + 8] = hi[i]; }
  return r;
}

__device__ __forceinline__ v8f wmma_bf16(v16bf a, v16bf b, v8f c) {
  return __builtin_amdgcn_wmma_f32_16x16x32_bf16(false, a, false, b, (short)0, c,
                                                 false, false);
}

// ---------------------------------------------------------------------------
// Generic bf16 WMMA GEMM:  C[M,N] = A[M,K] * W[N,K]^T + bias1 + bias2
// One wave -> 16x64 C tile (4 accumulators share the A fragment).
// Fast path (n0+64 <= N) is branch-free, K unrolled by 2 (all K % 64 == 0)
// so the second chunk's loads overlap the first chunk's WMMAs.
// Edge path (pred's last 16 columns) loops tiles singly under scalar branches.
// grid = (ceil(N/256), M/16), block = 128 (4 waves)
// ---------------------------------------------------------------------------
__global__ __launch_bounds__(128) void k_gemm(
    const bf16_t* __restrict__ Amat, int lda,
    const bf16_t* __restrict__ Wmat, int K,
    const float* __restrict__ bias1, const float* __restrict__ bias2,
    float* __restrict__ Cmat, long long ldc,
    int M, int N, const int* __restrict__ mask_declen, int t)
{
  const int lane = threadIdx.x & 31;
  // force wave id scalar so n0 / tile guards are SGPRs -> scalar branches,
  // EXEC stays all-ones around every WMMA (ISA requirement).
  const int wave = __builtin_amdgcn_readfirstlane((int)(threadIdx.x >> 5));
  const int m0 = blockIdx.y << 4;
  const int n0 = (blockIdx.x * 4 + wave) << 6;
  if (n0 >= N) return;

  const int l15  = lane & 15;
  const int koff = (lane & 16) ? 8 : 0;
  const int msub = (lane & 16) ? 8 : 0;

  const bf16_t* Ap  = Amat + (size_t)(m0 + l15) * lda + koff;
  const bf16_t* Wp  = Wmat + (size_t)(n0 + l15) * K + koff;
  const size_t  wst = (size_t)16 * K;   // stride between 16-col W tiles

  if (n0 + 64 <= N) {
    // ---------------- fast path: full 16x64 tile, no guards ---------------
    v8f a0 = {}, a1 = {}, a2 = {}, a3 = {};
    for (int k0 = 0; k0 < K; k0 += 64) {
      // prefetch weight stream a few chunks ahead (global_prefetch_b8)
      if (k0 + 256 < K) {
        __builtin_prefetch(Wp + 0 * wst + k0 + 256, 0, 3);
        __builtin_prefetch(Wp + 2 * wst + k0 + 256, 0, 3);
      }
      v16bf afA = load_frag(Ap + k0);
      v16bf b0A = load_frag(Wp + 0 * wst + k0);
      v16bf b1A = load_frag(Wp + 1 * wst + k0);
      v16bf b2A = load_frag(Wp + 2 * wst + k0);
      v16bf b3A = load_frag(Wp + 3 * wst + k0);
      v16bf afB = load_frag(Ap + k0 + 32);
      v16bf b0B = load_frag(Wp + 0 * wst + k0 + 32);
      v16bf b1B = load_frag(Wp + 1 * wst + k0 + 32);
      v16bf b2B = load_frag(Wp + 2 * wst + k0 + 32);
      v16bf b3B = load_frag(Wp + 3 * wst + k0 + 32);
      a0 = wmma_bf16(afA, b0A, a0);
      a1 = wmma_bf16(afA, b1A, a1);
      a2 = wmma_bf16(afA, b2A, a2);
      a3 = wmma_bf16(afA, b3A, a3);
      a0 = wmma_bf16(afB, b0B, a0);
      a1 = wmma_bf16(afB, b1B, a1);
      a2 = wmma_bf16(afB, b2B, a2);
      a3 = wmma_bf16(afB, b3B, a3);
    }
    // epilogue: VGPR r -> M = m0+r (+8 for lanes 16-31), N = n0+j*16+l15
    v8f accs[4] = {a0, a1, a2, a3};
#pragma unroll
    for (int j = 0; j < 4; ++j) {
      const int n = n0 + j * 16 + l15;
      const float bsum = (bias1 ? bias1[n] : 0.0f) + (bias2 ? bias2[n] : 0.0f);
#pragma unroll
      for (int r = 0; r < 8; ++r) {
        const int m = m0 + msub + r;
        float v = accs[j][r] + bsum;
        if (mask_declen) v = (t < mask_declen[m]) ? v : 0.0f;
        Cmat[(size_t)m * ldc + n] = v;
      }
    }
  } else {
    // ---------------- edge path: 1..3 16-col tiles (scalar-branched) -------
    const int jmax = (N - n0) >> 4;
    for (int j = 0; j < jmax; ++j) {
      v8f acc = {};
      const bf16_t* Wj = Wp + (size_t)j * wst;
      for (int k0 = 0; k0 < K; k0 += 64) {
        v16bf afA = load_frag(Ap + k0);
        v16bf bA  = load_frag(Wj + k0);
        v16bf afB = load_frag(Ap + k0 + 32);
        v16bf bB  = load_frag(Wj + k0 + 32);
        acc = wmma_bf16(afA, bA, acc);
        acc = wmma_bf16(afB, bB, acc);
      }
      const int n = n0 + j * 16 + l15;
      const float bsum = (bias1 ? bias1[n] : 0.0f) + (bias2 ? bias2[n] : 0.0f);
#pragma unroll
      for (int r = 0; r < 8; ++r) {
        const int m = m0 + msub + r;
        float v = acc[r] + bsum;
        if (mask_declen) v = (t < mask_declen[m]) ? v : 0.0f;
        Cmat[(size_t)m * ldc + n] = v;
      }
    }
  }
}

// ---------------------------------------------------------------------------
// Stable argsort of -lens (rank counting), emit positions, dec_len, and the
// integer tail outputs (as floats): sequences[pos], dec_len, positions.
// 1 block, 128 threads.
// ---------------------------------------------------------------------------
__global__ void k_sort(const int* __restrict__ sizes, const int* __restrict__ sequences,
                       int* __restrict__ pos, int* __restrict__ declen,
                       float* __restrict__ out_tail)
{
  __shared__ int spos[B_];
  const int i = threadIdx.x;
  const int li = sizes[i];
  int rank = 0;
  for (int j = 0; j < B_; ++j) {
    const int lj = sizes[j];
    if (lj > li || (lj == li && j < i)) rank++;
  }
  spos[rank] = i;
  __syncthreads();
  const int p = spos[i];
  pos[i] = p;
  const int dl = sizes[p] - 1;
  declen[i] = dl;
  for (int j = 0; j < ML_; ++j) out_tail[i * ML_ + j] = (float)sequences[p * ML_ + j];
  out_tail[B_ * ML_ + i]       = (float)dl;
  out_tail[B_ * ML_ + B_ + i]  = (float)p;
}

// permuted feats -> bf16
__global__ void k_permute_feats(const float* __restrict__ feats, const int* __restrict__ pos,
                                bf16_t* __restrict__ out)
{
  const size_t idx = (size_t)blockIdx.x * blockDim.x + threadIdx.x;
  if (idx >= (size_t)B_ * R_ * F_) return;
  const int b = (int)(idx / (R_ * F_));
  const size_t rem = idx % (R_ * F_);
  out[idx] = (bf16_t)feats[(size_t)pos[b] * R_ * F_ + rem];
}

// permuted embedding gather -> bf16 : (B, ML, E)
__global__ void k_gather_emb(const float* __restrict__ emb, const int* __restrict__ sequences,
                             const int* __restrict__ pos, bf16_t* __restrict__ out)
{
  const size_t idx = (size_t)blockIdx.x * blockDim.x + threadIdx.x;
  if (idx >= (size_t)B_ * ML_ * E_) return;
  const int b  = (int)(idx / (ML_ * E_));
  const int r2 = (int)(idx % (ML_ * E_));
  const int tt = r2 / E_;
  const int e  = r2 % E_;
  const int tok = sequences[pos[b] * ML_ + tt];
  out[idx] = (bf16_t)emb[(size_t)tok * E_ + e];
}

// generic f32 -> bf16
__global__ void k_convert(const float* __restrict__ in, bf16_t* __restrict__ out, size_t n)
{
  const size_t idx = (size_t)blockIdx.x * blockDim.x + threadIdx.x;
  if (idx < n) out[idx] = (bf16_t)in[idx];
}

// Wg[n][0:4096] = [W_ih[n][0:3072] | W_hh[n][0:1024]]  (bf16)
__global__ void k_concat_wg(const float* __restrict__ W_ih, const float* __restrict__ W_hh,
                            bf16_t* __restrict__ out)
{
  const size_t idx = (size_t)blockIdx.x * blockDim.x + threadIdx.x;
  if (idx >= (size_t)NG_ * KG_) return;
  const int n = (int)(idx / KG_);
  const int k = (int)(idx % KG_);
  const float v = (k < F_ + E_) ? W_ih[(size_t)n * (F_ + E_) + k]
                                : W_hh[(size_t)n * D_ + (k - (F_ + E_))];
  out[idx] = (bf16_t)v;
}

// Wp[n][0:3072] = [Wh[n][0:1024] | Wi[n][0:2048]]  (bf16)
__global__ void k_concat_wp(const float* __restrict__ Wh, const float* __restrict__ Wi,
                            bf16_t* __restrict__ out)
{
  const size_t idx = (size_t)blockIdx.x * blockDim.x + threadIdx.x;
  if (idx >= (size_t)V_ * KP_) return;
  const int n = (int)(idx / KP_);
  const int k = (int)(idx % KP_);
  const float v = (k < D_) ? Wh[(size_t)n * D_ + k]
                           : Wi[(size_t)n * F_ + (k - D_)];
  out[idx] = (bf16_t)v;
}

// zero c and the h slice of xcat (h0 = c0 = 0)
__global__ void k_init(float* __restrict__ c, bf16_t* __restrict__ xcat)
{
  const int idx = blockIdx.x * blockDim.x + threadIdx.x;
  if (idx >= B_ * D_) return;
  const int b = idx / D_, d = idx % D_;
  c[idx] = 0.0f;
  xcat[(size_t)b * KG_ + (F_ + E_) + d] = (bf16_t)0.0f;
}

// scores[b][r] = Wa . relu(h_att[b] + img_att[b,r]) + ba ; block per (b,r)
__global__ void k_scores(const float* __restrict__ hatt, const float* __restrict__ img_att,
                         const float* __restrict__ Wa, const float* __restrict__ ba,
                         float* __restrict__ scores)
{
  const int b = blockIdx.x, r = blockIdx.y;
  __shared__ float red[256];
  const float* ha = hatt + (size_t)b * A_;
  const float* ia = img_att + ((size_t)b * R_ + r) * A_;
  float s = 0.0f;
  for (int a = threadIdx.x; a < A_; a += blockDim.x) {
    float v = ha[a] + ia[a];
    v = v > 0.0f ? v : 0.0f;
    s += v * Wa[a];
  }
  red[threadIdx.x] = s;
  __syncthreads();
  for (int st = 128; st > 0; st >>= 1) {
    if ((int)threadIdx.x < st) red[threadIdx.x] += red[threadIdx.x + st];
    __syncthreads();
  }
  if (threadIdx.x == 0) scores[b * R_ + r] = red[0] + ba[0];
}

// softmax over R, aw = feats^T alpha (f32, gathered through positions);
// also writes bf16 aw into xcat[:,0:2048] and z[:,1024:3072], and copies
// embeds[:,t,:] into xcat[:,2048:3072]. block per b.
__global__ void k_softmax_aw(const float* __restrict__ scores, const float* __restrict__ feats,
                             const int* __restrict__ pos, const bf16_t* __restrict__ embeds,
                             float* __restrict__ aw, bf16_t* __restrict__ xcat,
                             bf16_t* __restrict__ z, int t)
{
  const int b = blockIdx.x;
  __shared__ float alpha[R_];
  if (threadIdx.x == 0) {
    float mx = -1e30f;
    for (int r = 0; r < R_; ++r) mx = fmaxf(mx, scores[b * R_ + r]);
    float s = 0.0f;
    for (int r = 0; r < R_; ++r) { float e = __expf(scores[b * R_ + r] - mx); alpha[r] = e; s += e; }
    const float inv = 1.0f / s;
    for (int r = 0; r < R_; ++r) alpha[r] *= inv;
  }
  __syncthreads();
  const float* fb = feats + (size_t)pos[b] * R_ * F_;
  for (int f = threadIdx.x; f < F_; f += blockDim.x) {
    float acc = 0.0f;
#pragma unroll
    for (int r = 0; r < R_; ++r) acc += alpha[r] * fb[(size_t)r * F_ + f];
    aw[(size_t)b * F_ + f] = acc;
    const bf16_t ab = (bf16_t)acc;
    xcat[(size_t)b * KG_ + f] = ab;
    z[(size_t)b * KP_ + D_ + f] = ab;
  }
  for (int e = threadIdx.x; e < E_; e += blockDim.x) {
    xcat[(size_t)b * KG_ + F_ + e] = embeds[((size_t)b * ML_ + t) * E_ + e];
  }
}

// LSTM pointwise; writes new h (bf16) into xcat h-slice (for next step) and z.
__global__ void k_lstm(const float* __restrict__ gates, float* __restrict__ c,
                       bf16_t* __restrict__ xcat, bf16_t* __restrict__ z)
{
  const int idx = blockIdx.x * blockDim.x + threadIdx.x;
  if (idx >= B_ * D_) return;
  const int b = idx / D_, d = idx % D_;
  const float* g = gates + (size_t)b * NG_;
  const float ig = sigmoidf_(g[d]);
  const float fg = sigmoidf_(g[D_ + d]);
  const float gg = tanhf(g[2 * D_ + d]);
  const float og = sigmoidf_(g[3 * D_ + d]);
  const float cn = fg * c[idx] + ig * gg;
  const float hn = og * tanhf(cn);
  c[idx] = cn;
  const bf16_t hb = (bf16_t)hn;
  xcat[(size_t)b * KG_ + (F_ + E_) + d] = hb;
  z[(size_t)b * KP_ + d] = hb;
}

// ---------------------------------------------------------------------------
// Host driver
// ---------------------------------------------------------------------------
extern "C" void kernel_launch(void* const* d_in, const int* in_sizes, int n_in,
                              void* d_out, int out_size, void* d_ws, size_t ws_size,
                              hipStream_t stream)
{
  (void)in_sizes; (void)n_in; (void)out_size; (void)ws_size;

  const float* feats     = (const float*)d_in[0];
  const int*   sequences = (const int*)d_in[1];
  const int*   sizes     = (const int*)d_in[2];
  const float* emb       = (const float*)d_in[3];
  const float* Wf        = (const float*)d_in[4];
  const float* bfv       = (const float*)d_in[5];
  const float* Wd        = (const float*)d_in[6];
  const float* bd        = (const float*)d_in[7];
  const float* Wa        = (const float*)d_in[8];
  const float* ba        = (const float*)d_in[9];
  const float* W_ih      = (const float*)d_in[10];
  const float* b_ih      = (const float*)d_in[11];
  const float* W_hh      = (const float*)d_in[12];
  const float* b_hh      = (const float*)d_in[13];
  const float* Wh        = (const float*)d_in[14];
  const float* bh        = (const float*)d_in[15];
  const float* Wi        = (const float*)d_in[16];
  const float* bi        = (const float*)d_in[17];

  float* out      = (float*)d_out;
  float* out_tail = out + (size_t)B_ * T_ * V_;

  // workspace carve-up
  char* ws = (char*)d_ws;
  size_t off = 0;
  auto wsa = [&](size_t bytes) -> void* {
    void* p = ws + off; off += (bytes + 255) & ~(size_t)255; return p;
  };
  int*    pos     = (int*)   wsa((size_t)B_ * 4);
  int*    declen  = (int*)   wsa((size_t)B_ * 4);
  bf16_t* featsbf = (bf16_t*)wsa((size_t)B_ * R_ * F_ * 2);
  bf16_t* embbf   = (bf16_t*)wsa((size_t)B_ * ML_ * E_ * 2);
  bf16_t* Wfb     = (bf16_t*)wsa((size_t)A_ * F_ * 2);
  bf16_t* Wdb     = (bf16_t*)wsa((size_t)A_ * D_ * 2);
  bf16_t* Wgb     = (bf16_t*)wsa((size_t)NG_ * KG_ * 2);
  bf16_t* Wpb     = (bf16_t*)wsa((size_t)V_ * KP_ * 2);
  float*  img_att = (float*) wsa((size_t)B_ * R_ * A_ * 4);
  float*  hatt    = (float*) wsa((size_t)B_ * A_ * 4);
  float*  scores  = (float*) wsa((size_t)B_ * R_ * 4);
  float*  aw      = (float*) wsa((size_t)B_ * F_ * 4);
  bf16_t* xcat    = (bf16_t*)wsa((size_t)B_ * KG_ * 2);
  bf16_t* z       = (bf16_t*)wsa((size_t)B_ * KP_ * 2);
  float*  gates   = (float*) wsa((size_t)B_ * NG_ * 4);
  float*  cbuf    = (float*) wsa((size_t)B_ * D_ * 4);

  const int TPB = 256;
  auto nb = [](size_t n, int tpb) { return (unsigned)((n + tpb - 1) / tpb); };

  // ---- one-time preprocessing ----
  k_sort<<<1, B_, 0, stream>>>(sizes, sequences, pos, declen, out_tail);
  k_permute_feats<<<nb((size_t)B_ * R_ * F_, TPB), TPB, 0, stream>>>(feats, pos, featsbf);
  k_gather_emb<<<nb((size_t)B_ * ML_ * E_, TPB), TPB, 0, stream>>>(emb, sequences, pos, embbf);
  k_convert<<<nb((size_t)A_ * F_, TPB), TPB, 0, stream>>>(Wf, Wfb, (size_t)A_ * F_);
  k_convert<<<nb((size_t)A_ * D_, TPB), TPB, 0, stream>>>(Wd, Wdb, (size_t)A_ * D_);
  k_concat_wg<<<nb((size_t)NG_ * KG_, TPB), TPB, 0, stream>>>(W_ih, W_hh, Wgb);
  k_concat_wp<<<nb((size_t)V_ * KP_, TPB), TPB, 0, stream>>>(Wh, Wi, Wpb);
  k_init<<<nb((size_t)B_ * D_, TPB), TPB, 0, stream>>>(cbuf, xcat);

  // img_att = featsP @ Wf^T + bf : M = B*R = 4608, N = 512, K = 2048
  {
    dim3 g((A_ + 255) / 256, (B_ * R_) / 16);
    k_gemm<<<g, 128, 0, stream>>>(featsbf, F_, Wfb, F_, bfv, nullptr,
                                  img_att, A_, B_ * R_, A_, nullptr, 0);
  }

  // ---- recurrence ----
  for (int t = 0; t < T_; ++t) {
    // h_att = h @ Wd^T + bd : A slice of xcat (h), M=128, N=512, K=1024
    {
      dim3 g((A_ + 255) / 256, B_ / 16);
      k_gemm<<<g, 128, 0, stream>>>(xcat + (F_ + E_), KG_, Wdb, D_, bd, nullptr,
                                    hatt, A_, B_, A_, nullptr, 0);
    }
    {
      dim3 g(B_, R_);
      k_scores<<<g, 256, 0, stream>>>(hatt, img_att, Wa, ba, scores);
    }
    k_softmax_aw<<<B_, 256, 0, stream>>>(scores, feats, pos, embbf, aw, xcat, z, t);

    // gates = xcat @ Wg^T (+ b_ih + b_hh) : M=128, N=4096, K=4096
    {
      dim3 g((NG_ + 255) / 256, B_ / 16);
      k_gemm<<<g, 128, 0, stream>>>(xcat, KG_, Wgb, KG_, b_ih, b_hh,
                                    gates, NG_, B_, NG_, nullptr, 0);
    }
    k_lstm<<<nb((size_t)B_ * D_, TPB), TPB, 0, stream>>>(gates, cbuf, xcat, z);

    // pred = z @ Wp^T + bh + bi, masked, written straight into d_out[b][t][:]
    {
      dim3 g((V_ + 255) / 256, B_ / 16);
      k_gemm<<<g, 128, 0, stream>>>(z, KP_, Wpb, KP_, bh, bi,
                                    out + (size_t)t * V_, (long long)T_ * V_,
                                    B_, V_, declen, t);
    }
  }
}
